// SchNetLayer_73993696575522
// MI455X (gfx1250) — matrix operand order repaired and checked by reading, not compiled, not run on hardware
//
#include <hip/hip_runtime.h>
#include <hip/hip_bf16.h>
#include <math.h>

// ---------------------------------------------------------------------------
// SchNet layer for MI455X (gfx1250, wave32).
// GEMM work (29+ GFLOP) runs on v_wmma_f32_16x16x32_bf16 (bf16 A/B, f32 acc).
// h (25.6MB) is L2-resident (192MB L2); edge gather + 82M fp32 scatter-adds
// hit L2, HBM traffic ~100MB -> the kernel is L2-atomic bound, so WMMA math
// is essentially free and we keep fp32 accumulation throughout.
// ---------------------------------------------------------------------------

#define N_NODES  50000
#define N_EDGES  640000
#define HIDDEN   128
#define NUM_RBF  50
#define CUTOFF_F 5.0f
#define WAVES    8          // waves per block (256 threads)

typedef __attribute__((ext_vector_type(16))) __bf16 v16bf;
typedef __attribute__((ext_vector_type(8)))  __bf16 v8bf;
typedef __attribute__((ext_vector_type(8)))  float  v8f;

// D = A(16x32 bf16) * B(32x16 bf16) + C(16x16 f32)
__device__ inline v8f wmma_bf16(v16bf a, v16bf b, v8f c) {
    return __builtin_amdgcn_wmma_f32_16x16x32_bf16(
        /*neg_a=*/false, a, /*neg_b=*/false, b,
        /*c_mod=*/(short)0, c, /*reuse_a=*/false, /*reuse_b=*/false);
}

// Assemble a 16x32 bf16 A-fragment from a row-major [16][rowStride] bf16 tile.
// ISA 7.12.2 (16-bit A 16x32): lane holds M=lane&15;
//   elements 0..7  <-> K = kc*32 + (lane>=16?8:0) + e
//   elements 8..15 <-> K = kc*32 + (lane>=16?8:0) + 16 + (e-8)
// i.e. two contiguous 16B runs at +0 and +16 elements from the lane base.
__device__ inline v16bf make_afrag(const __bf16* laneBase) {
    v8bf lo = *(const v8bf*)(laneBase);
    v8bf hi = *(const v8bf*)(laneBase + 16);
    v16bf a;
#pragma unroll
    for (int i = 0; i < 8; ++i) { a[i] = lo[i]; a[i + 8] = hi[i]; }
    return a;
}

// B fragments are pre-packed so each lane loads its 16 bf16 (32B) contiguously.
__device__ inline v16bf load_bfrag(const __bf16* packed, int kc, int nb, int lane) {
    return *(const v16bf*)(packed + (((size_t)(kc * 8 + nb) * 32 + lane) * 16));
}

__device__ inline float silu_f(float v) {
    return __fdividef(v, 1.0f + __expf(-v));
}

// ----------------------------- prep kernels --------------------------------

__global__ void zero_f32_kernel(float* __restrict__ p, int n) {
    int i = blockIdx.x * blockDim.x + threadIdx.x;
    if (i < n) p[i] = 0.0f;
}

// Repack row-major fp32 weight W[Ksrc][128] (zero-padded to nkc*32 rows) into
// fragment-native bf16: dst[((kc*8+nb)*32+lane)*16+e] = W[k][n] with
// n = nb*16 + (lane&15), k = kc*32 + ((lane>=16)?8:0) + (e<8 ? e : e+8)   (ISA B layout).
__global__ void pack_weights_kernel(const float* __restrict__ W,
                                    __bf16* __restrict__ dst,
                                    int Ksrc, int nkc) {
    int idx = blockIdx.x * blockDim.x + threadIdx.x;
    int total = nkc * 8 * 32 * 16;
    if (idx >= total) return;
    int e    = idx & 15;
    int lane = (idx >> 4) & 31;
    int nb   = (idx >> 9) & 7;
    int kc   = idx >> 12;
    int n = nb * 16 + (lane & 15);
    int k = kc * 32 + ((lane >= 16) ? 8 : 0) + ((e < 8) ? e : e + 8);
    float v = (k < Ksrc) ? W[(size_t)k * HIDDEN + n] : 0.0f;
    dst[idx] = (__bf16)v;
}

// ----------------------------- edge kernel ---------------------------------
// One wave per 16-edge tile. RBF(16x64 padded) @ fw1(64x128) -> SiLU
// -> @ fw2(128x128) -> *cutoff -> msg = h[col]*w -> atomicAdd agg[row].

__global__ __launch_bounds__(WAVES * 32)
void schnet_edge_kernel(const float* __restrict__ h,
                        const float* __restrict__ pos,
                        const int*   __restrict__ ei,      // [2, N_EDGES]
                        const __bf16* __restrict__ fw1f,
                        const float* __restrict__ fb1,
                        const __bf16* __restrict__ fw2f,
                        const float* __restrict__ fb2,
                        float* __restrict__ agg,
                        int ntiles, int nedges) {
    __shared__ __bf16 s_rbf[WAVES][16 * 64];
    __shared__ __bf16 s_x  [WAVES][16 * 128];
    __shared__ float  s_dist[WAVES][16];
    __shared__ float  s_cut [WAVES][16];
    __shared__ int    s_row [WAVES][16];
    __shared__ int    s_col [WAVES][16];

    const int wave = threadIdx.x >> 5;
    const int lane = threadIdx.x & 31;
    const int tile = blockIdx.x * WAVES + wave;
    const bool active = tile < ntiles;            // wave-uniform
    const int e0 = tile * 16;
    const int laneN = lane & 15;
    const int mhi = (lane >= 16) ? 8 : 0;

    if (active && lane < 16) {
        int e = e0 + lane;
        int r = ei[e];
        int c = ei[nedges + e];
        float dx = pos[r * 3 + 0] - pos[c * 3 + 0];
        float dy = pos[r * 3 + 1] - pos[c * 3 + 1];
        float dz = pos[r * 3 + 2] - pos[c * 3 + 2];
        float d = sqrtf(dx * dx + dy * dy + dz * dz + 1e-8f);
        s_row[wave][lane] = r;
        s_col[wave][lane] = c;
        s_dist[wave][lane] = d;
        float cut = 0.5f * (__cosf(3.14159265358979f * d / CUTOFF_F) + 1.0f);
        s_cut[wave][lane] = (d <= CUTOFF_F) ? cut : 0.0f;
    }
    __syncthreads();

    if (active) {
        // Gaussian RBF into 16x64 bf16 tile (K padded 50 -> 64 with zeros).
        const float inv_w = (float)(NUM_RBF - 1) / CUTOFF_F;   // 1/width
        for (int i = lane; i < 16 * 64; i += 32) {
            int m = i >> 6, k = i & 63;
            float val = 0.0f;
            if (k < NUM_RBF) {
                float t = s_dist[wave][m] * inv_w - (float)k;
                val = __expf(-0.5f * t * t);
            }
            s_rbf[wave][i] = (__bf16)val;
        }
    }
    __syncthreads();

    if (active) {
        // GEMM1: [16x64] @ [64x128]
        const __bf16* arow = &s_rbf[wave][(lane & 15) * 64 + ((lane >= 16) ? 8 : 0)];
        v16bf a0 = make_afrag(arow);
        v16bf a1 = make_afrag(arow + 32);
#pragma unroll
        for (int nb = 0; nb < 8; ++nb) {
            v8f acc = {};
            acc = wmma_bf16(a0, load_bfrag(fw1f, 0, nb, lane), acc);
            acc = wmma_bf16(a1, load_bfrag(fw1f, 1, nb, lane), acc);
            int n = nb * 16 + laneN;
            float bias = fb1[n];
#pragma unroll
            for (int r = 0; r < 8; ++r) {
                float v = silu_f(acc[r] + bias);
                s_x[wave][(r + mhi) * 128 + n] = (__bf16)v;   // transpose via LDS
            }
        }
    }
    __syncthreads();

    if (active) {
        // GEMM2: [16x128] @ [128x128], then cutoff * h[col] gather + scatter-add.
        const __bf16* arow = &s_x[wave][(lane & 15) * 128 + ((lane >= 16) ? 8 : 0)];
        v16bf a[4];
#pragma unroll
        for (int kc = 0; kc < 4; ++kc) a[kc] = make_afrag(arow + kc * 32);

        int rowm[8], colm[8]; float cutm[8];
#pragma unroll
        for (int r = 0; r < 8; ++r) {
            int m = r + mhi;
            rowm[r] = s_row[wave][m];
            colm[r] = s_col[wave][m];
            cutm[r] = s_cut[wave][m];
        }
#pragma unroll
        for (int nb = 0; nb < 8; ++nb) {
            v8f acc = {};
#pragma unroll
            for (int kc = 0; kc < 4; ++kc)
                acc = wmma_bf16(a[kc], load_bfrag(fw2f, kc, nb, lane), acc);
            int n = nb * 16 + laneN;
            float b2 = fb2[n];
#pragma unroll
            for (int r = 0; r < 8; ++r) {
                float w = (acc[r] + b2) * cutm[r];
                float msg = h[(size_t)colm[r] * HIDDEN + n] * w;
                atomicAdd(&agg[(size_t)rowm[r] * HIDDEN + n], msg);
            }
        }
    }
}

// ----------------------------- node kernel ---------------------------------
// out = h + silu(agg @ iw1 + ib1) @ iw2 + ib2, one wave per 16-node tile.

__global__ __launch_bounds__(WAVES * 32)
void schnet_node_kernel(const float* __restrict__ h,
                        const float* __restrict__ agg,
                        const __bf16* __restrict__ iw1f,
                        const float* __restrict__ ib1,
                        const __bf16* __restrict__ iw2f,
                        const float* __restrict__ ib2,
                        float* __restrict__ out,
                        int ntiles) {
    __shared__ __bf16 s_x[WAVES][16 * 128];

    const int wave = threadIdx.x >> 5;
    const int lane = threadIdx.x & 31;
    const int tile = blockIdx.x * WAVES + wave;
    const bool active = tile < ntiles;            // wave-uniform
    const int t0 = tile * 16;
    const int laneN = lane & 15;
    const int mhi = (lane >= 16) ? 8 : 0;

    if (active) {
        for (int i = lane; i < 16 * 128; i += 32) {
            int m = i >> 7, k = i & 127;
            s_x[wave][i] = (__bf16)agg[(size_t)(t0 + m) * HIDDEN + k];
        }
    }
    __syncthreads();

    if (active) {
        const __bf16* arow = &s_x[wave][(lane & 15) * 128 + ((lane >= 16) ? 8 : 0)];
        v16bf a[4];
#pragma unroll
        for (int kc = 0; kc < 4; ++kc) a[kc] = make_afrag(arow + kc * 32);
#pragma unroll
        for (int nb = 0; nb < 8; ++nb) {
            v8f acc = {};
#pragma unroll
            for (int kc = 0; kc < 4; ++kc)
                acc = wmma_bf16(a[kc], load_bfrag(iw1f, kc, nb, lane), acc);
            int n = nb * 16 + laneN;
            float b1 = ib1[n];
#pragma unroll
            for (int r = 0; r < 8; ++r) {
                float v = silu_f(acc[r] + b1);
                s_x[wave][(r + mhi) * 128 + n] = (__bf16)v;   // reuse tile (A already in regs)
            }
        }
    }
    __syncthreads();

    if (active) {
        const __bf16* arow = &s_x[wave][(lane & 15) * 128 + ((lane >= 16) ? 8 : 0)];
        v16bf a[4];
#pragma unroll
        for (int kc = 0; kc < 4; ++kc) a[kc] = make_afrag(arow + kc * 32);
#pragma unroll
        for (int nb = 0; nb < 8; ++nb) {
            v8f acc = {};
#pragma unroll
            for (int kc = 0; kc < 4; ++kc)
                acc = wmma_bf16(a[kc], load_bfrag(iw2f, kc, nb, lane), acc);
            int n = nb * 16 + laneN;
            float b2 = ib2[n];
#pragma unroll
            for (int r = 0; r < 8; ++r) {
                size_t node = (size_t)(t0 + r + mhi);
                out[node * HIDDEN + n] = h[node * HIDDEN + n] + acc[r] + b2;
            }
        }
    }
}

// ----------------------------- launch --------------------------------------

extern "C" void kernel_launch(void* const* d_in, const int* in_sizes, int n_in,
                              void* d_out, int out_size, void* d_ws, size_t ws_size,
                              hipStream_t stream) {
    const float* h   = (const float*)d_in[0];
    const float* pos = (const float*)d_in[1];
    const int*   ei  = (const int*)  d_in[2];   // integer inputs arrive as int32
    const float* fw1 = (const float*)d_in[3];
    const float* fb1 = (const float*)d_in[4];
    const float* fw2 = (const float*)d_in[5];
    const float* fb2 = (const float*)d_in[6];
    const float* iw1 = (const float*)d_in[7];
    const float* ib1 = (const float*)d_in[8];
    const float* iw2 = (const float*)d_in[9];
    const float* ib2 = (const float*)d_in[10];
    float* out = (float*)d_out;

    // Workspace layout: agg[50000*128] f32, then 4 fragment-packed bf16 weights.
    float*  agg  = (float*)d_ws;
    __bf16* fw1f = (__bf16*)(agg + (size_t)N_NODES * HIDDEN);
    __bf16* fw2f = fw1f + 2 * 8 * 32 * 16;      //  8192 elems
    __bf16* iw1f = fw2f + 4 * 8 * 32 * 16;      // 16384 elems
    __bf16* iw2f = iw1f + 4 * 8 * 32 * 16;

    const int nagg = N_NODES * HIDDEN;
    zero_f32_kernel<<<(nagg + 255) / 256, 256, 0, stream>>>(agg, nagg);
    pack_weights_kernel<<<(2 * 8 * 32 * 16 + 255) / 256, 256, 0, stream>>>(fw1, fw1f, NUM_RBF, 2);
    pack_weights_kernel<<<(4 * 8 * 32 * 16 + 255) / 256, 256, 0, stream>>>(fw2, fw2f, HIDDEN, 4);
    pack_weights_kernel<<<(4 * 8 * 32 * 16 + 255) / 256, 256, 0, stream>>>(iw1, iw1f, HIDDEN, 4);
    pack_weights_kernel<<<(4 * 8 * 32 * 16 + 255) / 256, 256, 0, stream>>>(iw2, iw2f, HIDDEN, 4);

    const int etiles = N_EDGES / 16;            // 40000, divisible by WAVES
    schnet_edge_kernel<<<etiles / WAVES, WAVES * 32, 0, stream>>>(
        h, pos, ei, fw1f, fb1, fw2f, fb2, agg, etiles, N_EDGES);

    const int vtiles = N_NODES / 16;            // 3125
    schnet_node_kernel<<<(vtiles + WAVES - 1) / WAVES, WAVES * 32, 0, stream>>>(
        h, agg, iw1f, ib1, iw2f, ib2, out, vtiles);
}
